// STMamba_53730040873393
// MI455X (gfx1250) — compile-verified, matrix operand order checked by reading, not run on hardware
//
#include <hip/hip_runtime.h>
#include <hip/hip_bf16.h>
#include <math.h>

typedef __attribute__((ext_vector_type(16))) _Float16 v16h;
typedef __attribute__((ext_vector_type(8)))  _Float16 v8h;
typedef __attribute__((ext_vector_type(8)))  float    v8f;
typedef __attribute__((ext_vector_type(4)))  unsigned int v4u;
typedef __attribute__((ext_vector_type(8)))  int      v8i;
typedef __attribute__((ext_vector_type(4)))  int      v4i;

#define EPSN 1e-5f
#define W2_ELEMS (21 * 224 * 9)   // 42336

// ---------------- compile-time spiral scan order (replicates _scan_order(11)) ---------
struct Spiral { int idx[121]; };
static constexpr Spiral mkSpiral() {
  Spiral s{}; int rr[121] = {}, cc[121] = {};
  int n = 0; const int cen = 5;
  rr[n] = cen; cc[n] = cen; n++;
  for (int k = 1; k <= cen; ++k) {
    for (int x = cen - (k - 1); x <= cen + k; ++x) { rr[n] = cen - k; cc[n] = x; n++; }
    for (int y = cen - (k - 1); y <= cen + k; ++y) { rr[n] = y; cc[n] = cen + k; n++; }
    for (int x = cen - k; x <= cen + k - 1; ++x)   { rr[n] = cen + k; cc[n] = x; n++; }
    for (int y = cen - k; y <= cen + k - 1; ++y)   { rr[n] = y; cc[n] = cen - k; n++; }
  }
  for (int i = 0; i < 121; ++i) s.idx[i] = rr[i] * 11 + cc[i];
  return s;
}
__constant__ Spiral SPIRAL = mkSpiral();

// ---------------- kernel params ---------------------------------------------------
struct P {
  const float *x, *w3, *b3, *g3, *be3, *m3, *v3;
  const float *w2, *b2, *g2, *be2, *m2, *v2;
  const float *pos, *cls, *g1, *be1;
  const float *Wb, *bb, *Wc, *bc, *Wd, *bd, *dp, *A_, *Wst, *bst, *stbias;
  const float *gl2, *bl2, *Wm1, *bm1, *Wm2, *bm2;
  const _Float16 *w2h;   // pre-transformed f16 weights in d_ws (TDM source)
  float *out;
  int use_tdm;
};

// small persistent LDS parameter block (zpad first: must stay 16B aligned)
struct SmallP {
  float zpad[16];                 // 64B zero pad for out-of-range WMMA operand lanes
  float w3[216];
  float sc3[8],  bi3[8];
  float sc2[24], bi2[24];
  float g1[24],  b1[24];
  float sqm[24], stg[24];
  float pooled[24];
  float mv[2];
  float hid[8];
};
static_assert(sizeof(SmallP) <= 1728, "SmallP too big");

// LDS memory map (bytes)
#define SM_WB_OFF   0        // _Float16 wB[9][21][224]  = 84672 B
#define SM_PAR_OFF  84672    // SmallP (<=1728 B)
#define SM_DYN_OFF  86400    // dynamic region, 70272 B
#define SM_BYTES    156672
// dyn phase A: xs f32[6750] @ +0 ; h1 f16[169*128] @ +27008 (ends 70272 exactly)
// dyn phase B: h2 f32[2541] @ +0 ; Wsm f32[1764] @ +10176 ; seq @ +17232 ;
//              xln @ +27480 ; Bx @ +37728 ; Cm @ +47976 ; As @ +58224 (ends 68472)

__device__ __forceinline__ float sigm(float v) { return 1.f / (1.f + expf(-v)); }

// ---- conv3d + BN3 + ReLU into f16 LDS, phase-specialized trip count ---------------
template <int ICOUNT>
__device__ __forceinline__ void conv3d_phase(const float* __restrict__ xs,
                                             _Float16* __restrict__ h1,
                                             const SmallP* sp, int ibase, int tid) {
  const int total = ICOUNT * 169;
  for (int t = tid; t < total; t += 256) {
    int pix = t / ICOUNT; int ii = t - pix * ICOUNT;   // ICOUNT is a constant -> cheap
    int y = pix / 13, xc = pix - y * 13;
    int i = ibase + ii;
    int c = i / 28, d = i - c * 28;
    const float* wp = &sp->w3[c * 27];
    float acc = 0.f;
#pragma unroll
    for (int kd = 0; kd < 3; ++kd)
#pragma unroll
      for (int ky = 0; ky < 3; ++ky)
#pragma unroll
        for (int kx = 0; kx < 3; ++kx)
          acc += xs[(d + kd) * 225 + (y + ky) * 15 + (xc + kx)] * wp[kd * 9 + ky * 3 + kx];
    float h = fmaxf(acc * sp->sc3[c] + sp->bi3[c], 0.f);
    h1[pix * 128 + ii] = (_Float16)h;
  }
}

// ---- WMMA GEMM accumulation for one K phase; NK 32-wide chunks, fully unrolled ----
template <int NK>
__device__ __forceinline__ void gemm_accum(const _Float16* __restrict__ h1,
                                           const _Float16* __restrict__ wB,
                                           const _Float16* __restrict__ zp,
                                           int ibase, int prow, int hi, int lane,
                                           v8f& acc0, v8f& acc1) {
  const bool av = (prow < 121);
  int oy = 0, ox = 0;
  if (av) { oy = prow / 11; ox = prow - oy * 11; }
  const int oc0 = lane & 15, oc1 = oc0 + 16;
  const bool bv = (oc1 < 21);
#pragma unroll
  for (int r = 0; r < 9; ++r) {
    const int ky = r / 3, kx = r - ky * 3;
    const int apix = (oy + ky) * 13 + (ox + kx);
    const _Float16* abase  = av ? (h1 + apix * 128) : zp;
    const _Float16* b0base = wB + (r * 21 + oc0) * 224 + ibase;
    const _Float16* b1base = bv ? (wB + (r * 21 + oc1) * 224 + ibase) : zp;
#pragma unroll
    for (int kc = 0; kc < NK; ++kc) {
      const int kl = kc * 32 + hi * 8;
      const v8h* ap  = (const v8h*)(av ? (abase + kl) : zp);   // invalid lanes -> zero pad
      const v8h* bp0 = (const v8h*)(b0base + kl);
      const v8h* bp1 = (const v8h*)(bv ? (b1base + kl) : zp);
      v8h alo = ap[0],  ahi2 = ap[2];     // K{0..7} and K{16..23} relative, 16B each
      v8h b0l = bp0[0], b0h  = bp0[2];
      v8h b1l = bp1[0], b1h  = bp1[2];
      v16h a   = __builtin_shufflevector(alo, ahi2, 0,1,2,3,4,5,6,7,8,9,10,11,12,13,14,15);
      v16h b0v = __builtin_shufflevector(b0l, b0h,  0,1,2,3,4,5,6,7,8,9,10,11,12,13,14,15);
      v16h b1v = __builtin_shufflevector(b1l, b1h,  0,1,2,3,4,5,6,7,8,9,10,11,12,13,14,15);
      acc0 = __builtin_amdgcn_wmma_f32_16x16x32_f16(false, a, false, b0v, (short)0, acc0, false, false);
      acc1 = __builtin_amdgcn_wmma_f32_16x16x32_f16(false, a, false, b1v, (short)0, acc1, false, false);
    }
  }
}

// ---- TDM: async copy of the 84672-byte f16 weight image into LDS (1-D tile) -------
// Toolchain probe: 6-arg builtin (g0 v4u, g1 v8i, g2 v4i, g3 v4i, g4 v8i, cpol)
__device__ __forceinline__ void tdm_load_wB(const _Float16* gsrc, unsigned lds_off) {
  unsigned long long ga = (unsigned long long)(uintptr_t)gsrc;
  const unsigned n = W2_ELEMS;                 // 42336 f16 elements (fits tile_dim0 u16)
  v4u g0;
  g0[0] = 1u;                                  // count=1 (valid user descriptor)
  g0[1] = lds_off;                             // lds_addr (bytes)
  g0[2] = (unsigned)(ga & 0xffffffffu);        // global_addr[31:0]
  g0[3] = (unsigned)((ga >> 32) & 0x01ffffffu) // global_addr[56:32]
        | (2u << 30);                          // type = 2 ("image")
  v8i g1;
  g1[0] = (int)(1u << 16);                     // data_size = 1 -> 2 bytes
  g1[1] = (int)((n & 0xffffu) << 16);          // tensor_dim0[15:0]
  g1[2] = (int)((n >> 16) | (1u << 16));       // tensor_dim0[31:16] | tensor_dim1=1
  g1[3] = (int)((n & 0xffffu) << 16);          // tile_dim0 = n
  g1[4] = 1;                                   // tile_dim1 = 1
  g1[5] = (int)n;                              // tensor_dim0_stride[31:0]
  g1[6] = 0;                                   // stride msbs / tensor_dim1_stride
  g1[7] = 0;
  v4i gz;  gz[0] = 0; gz[1] = 0; gz[2] = 0; gz[3] = 0;
  v8i gz8; gz8[0] = 0; gz8[1] = 0; gz8[2] = 0; gz8[3] = 0;
  gz8[4] = 0; gz8[5] = 0; gz8[6] = 0; gz8[7] = 0;
  __builtin_amdgcn_tensor_load_to_lds(g0, g1, gz, gz, gz8, 0);
}

// ---- pre-kernel: (O,I,ky,kx) f32 -> [r][o][i] f16 into d_ws -----------------------
__global__ __launch_bounds__(256)
void w2_to_f16(const float* __restrict__ w2, _Float16* __restrict__ dst) {
  for (int s = blockIdx.x * 256 + threadIdx.x; s < W2_ELEMS; s += gridDim.x * 256) {
    int o = s / 2016; int rem = s - o * 2016;
    int i = rem / 9;  int r = rem - i * 9;
    dst[(r * 21 + o) * 224 + i] = (_Float16)w2[s];
  }
}

__global__ __launch_bounds__(256)
void stmamba_fused(P p) {
  __shared__ __align__(16) unsigned char smem[SM_BYTES];
  _Float16*      wB  = (_Float16*)(smem + SM_WB_OFF);
  SmallP*        sp  = (SmallP*)(smem + SM_PAR_OFF);
  unsigned char* dyn = smem + SM_DYN_OFF;

  float*    xs  = (float*)(dyn);
  _Float16* h1  = (_Float16*)(dyn + 27008);
  float*    h2  = (float*)(dyn);
  float*    Wsm = (float*)(dyn + 10176);
  float*    seq = (float*)(dyn + 17232);
  float*    xln = (float*)(dyn + 27480);
  float*    Bx  = (float*)(dyn + 37728);
  float*    Cm  = (float*)(dyn + 47976);
  float*    As  = (float*)(dyn + 58224);

  const int b    = blockIdx.x;
  const int tid  = threadIdx.x;
  const int lane = tid & 31;
  const int wv   = tid >> 5;

  // ---- stage 0: cooperative loads -----------------------------------------------
  if (p.use_tdm) {
    if (wv == 0) {
      // async tensor DMA: contiguous 84672B f16 image -> LDS; wave-level op
      tdm_load_wB(p.w2h, (unsigned)(uintptr_t)(void*)wB + 0u);
      __builtin_amdgcn_s_wait_tensorcnt(0);
    }
  } else {
    for (int s = tid; s < W2_ELEMS; s += 256) {
      int o = s / 2016; int rem = s - o * 2016;
      int i = rem / 9;  int r = rem - i * 9;
      wB[(r * 21 + o) * 224 + i] = (_Float16)p.w2[s];
    }
  }
  const float* xg = p.x + (size_t)b * 6750;
  for (int s = tid; s < 6750; s += 256) xs[s] = xg[s];
  if (tid < 16) sp->zpad[tid] = 0.f;
  if (tid < 216) sp->w3[tid] = p.w3[tid];
  if (tid < 8) {
    float s3 = p.g3[tid] * rsqrtf(p.v3[tid] + EPSN);
    sp->sc3[tid] = s3;
    sp->bi3[tid] = (p.b3[tid] - p.m3[tid]) * s3 + p.be3[tid];
  }
  if (tid < 21) {
    float s2 = p.g2[tid] * rsqrtf(p.v2[tid] + EPSN);
    sp->sc2[tid] = s2;
    sp->bi2[tid] = (p.b2[tid] - p.m2[tid]) * s2 + p.be2[tid];
    sp->g1[tid]  = p.g1[tid];
    sp->b1[tid]  = p.be1[tid];
  }
  __syncthreads();

  // WMMA accumulators persist across the two K phases
  v8f acc0, acc1;
#pragma unroll
  for (int j = 0; j < 8; ++j) { acc0[j] = 0.f; acc1[j] = 0.f; }

  const int p0   = wv * 16;             // M-tile base row (output pixel)
  const int prow = p0 + (lane & 15);
  const int hi   = (lane >> 4) & 1;
  const _Float16* zp = (const _Float16*)sp->zpad;

  // ---- stage 1: phase 0 = channels 0..127, phase 1 = channels 128..223 -----------
  conv3d_phase<128>(xs, h1, sp, 0, tid);
  __syncthreads();
  gemm_accum<4>(h1, wB, zp, 0, prow, hi, lane, acc0, acc1);
  __syncthreads();
  conv3d_phase<96>(xs, h1, sp, 128, tid);
  __syncthreads();
  gemm_accum<3>(h1, wB, zp, 128, prow, hi, lane, acc0, acc1);
  __syncthreads();

  // ---- stage 2: BN2 + ReLU, spill GEMM result to LDS (h2 aliases dead xs) --------
  {
    const int oc0   = lane & 15;
    const int rbase = p0 + hi * 8;      // C-layout: lanes16-31 hold M = base+8..base+15
#pragma unroll
    for (int j = 0; j < 8; ++j) {
      int pr = rbase + j;
      if (pr < 121) {
        h2[pr * 21 + oc0] = fmaxf(acc0[j] * sp->sc2[oc0] + sp->bi2[oc0], 0.f);
        int oc1 = oc0 + 16;
        if (oc1 < 21)
          h2[pr * 21 + oc1] = fmaxf(acc1[j] * sp->sc2[oc1] + sp->bi2[oc1], 0.f);
      }
    }
  }
  for (int s = tid; s < 4 * 441; s += 256) {
    int m = s / 441; int j = s - m * 441;
    const float* src = (m == 0) ? p.Wb : (m == 1) ? p.Wc : (m == 2) ? p.Wd : p.Wst;
    Wsm[s] = src[j];
  }
  __syncthreads();

  // ---- stage 3: spiral gather + cls + pos_emb ------------------------------------
  for (int s = tid; s < 122 * 21; s += 256) {
    int row = s / 21, d = s - row * 21;
    float v = (row == 0) ? p.cls[d] : h2[SPIRAL.idx[row - 1] * 21 + d];
    seq[s] = v + p.pos[s];
  }
  __syncthreads();

  // ---- stage 4: LayerNorm(1), one thread per sequence row ------------------------
  if (tid < 122) {
    const float* rp = seq + tid * 21;
    float m = 0.f;
    for (int d = 0; d < 21; ++d) m += rp[d];
    m *= (1.f / 21.f);
    float v = 0.f;
    for (int d = 0; d < 21; ++d) { float t = rp[d] - m; v += t * t; }
    v *= (1.f / 21.f);
    float rs = rsqrtf(v + EPSN);
    float* op = xln + tid * 21;
    for (int d = 0; d < 21; ++d) op[d] = (rp[d] - m) * rs * sp->g1[d] + sp->b1[d];
  }
  __syncthreads();

  if (tid < 21) {            // column means of xln
    float ssum = 0.f;
    for (int row = 0; row < 122; ++row) ssum += xln[row * 21 + tid];
    sp->sqm[tid] = ssum * (1.f / 122.f);
  }
  __syncthreads();

  // ---- stage 5: gate vector + B0/C/delta pass ------------------------------------
  if (tid < 21) {
    const float* Wst = Wsm + 3 * 441;
    float a = 0.f;
    for (int k = 0; k < 21; ++k) a += sp->sqm[k] * Wst[k * 21 + tid];
    a = sigm(fmaxf(a + p.bst[tid], 0.f));
    sp->stg[tid] = a + p.stbias[tid];
  }
  for (int s = tid; s < 122 * 21; s += 256) {
    int row = s / 21, d = s - row * 21;
    const float* xr = xln + row * 21;
    const float* wb = Wsm;
    const float* wc = Wsm + 441;
    const float* wd = Wsm + 882;
    float ab = 0.f, ac = 0.f, ad = 0.f;
    for (int k = 0; k < 21; ++k) {
      float xv = xr[k];
      ab += xv * wb[k * 21 + d];
      ac += xv * wc[k * 21 + d];
      ad += xv * wd[k * 21 + d];
    }
    float B0    = ab + p.bb[d];
    float Cv    = ac + p.bc[d];
    float delta = sigm(ad + p.bd[d] + p.dp[s]);
    Cm[s] = Cv;
    Bx[s] = delta * B0 * xr[d];
    As[s] = delta * p.A_[s];
  }
  __syncthreads();

  // ---- stage 6: selective scan (21 independent 122-step recurrences) -------------
  if (tid < 21) {
    float sv = 0.f;
    for (int row = 0; row < 122; ++row) {
      int idx = row * 21 + tid;
      sv = As[idx] * sv + Bx[idx];
      As[idx] = sv;                 // in place: A -> S
    }
  }
  __syncthreads();

  // ---- stage 7: combine, residual, pool ------------------------------------------
  for (int s = tid; s < 122 * 21; s += 256) {
    int d  = s % 21;
    float xl = xln[s];
    float yv = (Cm[s] * As[s] + sp->stg[d] * xl) * sigm(xl);
    seq[s] = seq[s] + yv;
  }
  __syncthreads();
  if (tid < 21) {
    float ssum = 0.f;
    for (int row = 0; row < 122; ++row) ssum += seq[row * 21 + tid];
    sp->pooled[tid] = ssum * (1.f / 122.f);
  }
  __syncthreads();

  // ---- stage 8: LN2 + MLP head ----------------------------------------------------
  if (tid == 0) {
    float m = 0.f;
    for (int d = 0; d < 21; ++d) m += sp->pooled[d];
    m *= (1.f / 21.f);
    float v = 0.f;
    for (int d = 0; d < 21; ++d) { float t = sp->pooled[d] - m; v += t * t; }
    v *= (1.f / 21.f);
    sp->mv[0] = m; sp->mv[1] = rsqrtf(v + EPSN);
  }
  __syncthreads();
  if (tid < 8) {
    float m = sp->mv[0], rs = sp->mv[1];
    float a = p.bm1[tid];
    for (int d = 0; d < 21; ++d) {
      float pn = (sp->pooled[d] - m) * rs * p.gl2[d] + p.bl2[d];
      a += pn * p.Wm1[d * 8 + tid];
    }
    sp->hid[tid] = 0.5f * a * (1.f + erff(a * 0.70710678118654752f));
  }
  __syncthreads();
  if (tid < 21) {
    float a = p.bm2[tid];
    for (int j = 0; j < 8; ++j) a += sp->hid[j] * p.Wm2[j * 21 + tid];
    p.out[(size_t)b * 21 + tid] = sp->pooled[tid] + a;
  }
}

extern "C" void kernel_launch(void* const* d_in, const int* in_sizes, int n_in,
                              void* d_out, int out_size, void* d_ws, size_t ws_size,
                              hipStream_t stream) {
  (void)n_in; (void)out_size;
  P p;
  p.x      = (const float*)d_in[0];
  p.w3     = (const float*)d_in[1];
  p.b3     = (const float*)d_in[2];
  p.g3     = (const float*)d_in[3];
  p.be3    = (const float*)d_in[4];
  p.m3     = (const float*)d_in[5];
  p.v3     = (const float*)d_in[6];
  p.w2     = (const float*)d_in[7];
  p.b2     = (const float*)d_in[8];
  p.g2     = (const float*)d_in[9];
  p.be2    = (const float*)d_in[10];
  p.m2     = (const float*)d_in[11];
  p.v2     = (const float*)d_in[12];
  p.pos    = (const float*)d_in[13];
  p.cls    = (const float*)d_in[14];
  p.g1     = (const float*)d_in[15];
  p.be1    = (const float*)d_in[16];
  p.Wb     = (const float*)d_in[17];
  p.bb     = (const float*)d_in[18];
  p.Wc     = (const float*)d_in[19];
  p.bc     = (const float*)d_in[20];
  p.Wd     = (const float*)d_in[21];
  p.bd     = (const float*)d_in[22];
  p.dp     = (const float*)d_in[23];
  p.A_     = (const float*)d_in[24];
  p.Wst    = (const float*)d_in[25];
  p.bst    = (const float*)d_in[26];
  p.stbias = (const float*)d_in[27];
  p.gl2    = (const float*)d_in[28];
  p.bl2    = (const float*)d_in[29];
  p.Wm1    = (const float*)d_in[30];
  p.bm1    = (const float*)d_in[31];
  p.Wm2    = (const float*)d_in[32];
  p.bm2    = (const float*)d_in[33];
  p.out    = (float*)d_out;

  const bool tdm_ok = (d_ws != nullptr) && (ws_size >= (size_t)W2_ELEMS * sizeof(_Float16));
  p.w2h    = (const _Float16*)d_ws;
  p.use_tdm = tdm_ok ? 1 : 0;

  if (tdm_ok) {
    // pre-transform conv2d weights to the f16 [tap][o][i] image TDM copies into LDS
    hipLaunchKernelGGL(w2_to_f16, dim3(64), dim3(256), 0, stream,
                       (const float*)d_in[7], (_Float16*)d_ws);
  }
  const int B = in_sizes[0] / 6750;   // (B,1,30,15,15)
  hipLaunchKernelGGL(stmamba_fused, dim3(B), dim3(256), 0, stream, p);
}
